// LM_35983236006478
// MI455X (gfx1250) — matrix operand order repaired and checked by reading, not compiled, not run on hardware
//
#include <hip/hip_runtime.h>

#define DEV __device__ __forceinline__

typedef unsigned short u16;
typedef __attribute__((ext_vector_type(16))) __bf16    v16bf;
typedef __attribute__((ext_vector_type(8)))  float     v8f;
typedef __attribute__((ext_vector_type(4)))  unsigned  v4u;
typedef __attribute__((ext_vector_type(8)))  unsigned  v8u;

constexpr int D_  = 1024;
constexpr int NH_ = 16;
constexpr int NL_ = 8;
constexpr int L_  = 2048;
constexpr int V_  = 32000;
constexpr int B_  = 4;
constexpr int DH_ = 64;    // D_/NH_
constexpr int D3_ = 3072;  // 3*D_

#define NEG_INF (-__builtin_inff())

union FragU {
  uint4 q[2];
  v16bf v;
};

DEV u16 f2bf(float x) {
  unsigned u = __float_as_uint(x);
  unsigned r = u + 0x7FFFu + ((u >> 16) & 1u);  // round-to-nearest-even
  return (u16)(r >> 16);
}

DEV v8f v8zero() {
  v8f z = {0.f, 0.f, 0.f, 0.f, 0.f, 0.f, 0.f, 0.f};
  return z;
}

DEV v8f wmma_bf16(v16bf a, v16bf b, v8f c) {
  // D = A(16x32) * B(32x16) + C, fp32 accumulate
  return __builtin_amdgcn_wmma_f32_16x16x32_bf16(false, a, false, b, (short)0, c,
                                                 false, false);
}

// A-matrix fragment: 16x32 bf16 tile of X (row-major, leading dim ld), at (row0,k0).
// ISA 16-bit A layout: lanes 0-15 hold row M=lane, K in [0..7]+[16..23];
// lanes 16-31 hold row M=lane-16, K in [8..15]+[24..31].
DEV v16bf load_frag_a(const u16* X, int ld, int row0, int k0, int lane) {
  int hf = lane >> 4, m = lane & 15;
  const u16* p = X + (size_t)(row0 + m) * ld + k0 + hf * 8;
  FragU f;
  f.q[0] = *(const uint4*)(p);        // K chunk [ka .. ka+7]
  f.q[1] = *(const uint4*)(p + 16);   // K chunk [ka+16 .. ka+23]
  return f.v;
}

// B-matrix fragment for C = A * W^T where W is row-major N x ld (K-major rows).
// Dense 32x16 B layout: lanes 0-15 col N=lane with K=[k0..k0+7]+[k0+16..k0+23],
// lanes 16-31 with K=[k0+8..k0+15]+[k0+24..k0+31] (matches A layout).
DEV v16bf load_frag_b(const u16* W, int ld, int n0, int k0, int lane) {
  int hf = lane >> 4, n = lane & 15;
  const u16* p = W + (size_t)(n0 + n) * ld + k0 + hf * 8;
  FragU f;
  f.q[0] = *(const uint4*)(p);
  f.q[1] = *(const uint4*)(p + 16);
  return f.v;
}

// ---------------------------------------------------- CDNA5 async copy path

// Per-lane async global->LDS 16B copy; tracked by ASYNCcnt (not LOADcnt).
DEV void async_ld_b128(unsigned lds_byte_off, const char* gaddr) {
  asm volatile("global_load_async_to_lds_b128 %0, %1, off"
               :: "v"(lds_byte_off), "v"(gaddr)
               : "memory");
}

DEV void wait_async0() {
  asm volatile("s_wait_asynccnt 0" ::: "memory");
}

DEV void wait_tensor0() {
#if __has_builtin(__builtin_amdgcn_s_wait_tensorcnt)
  __builtin_amdgcn_s_wait_tensorcnt(0);
#else
  asm volatile("s_wait_tensorcnt 0" ::: "memory");
#endif
}

// Stage a 64-row x 32-col bf16 tile of X (row-major, ld elems) at (row0,k0)
// into LDS (64B rows). 128 threads x 2 chunks x 16B = 4KB.
DEV void issue_tile(const u16* X, int ld, int row0, int k0, u16* sh, int t) {
#pragma unroll
  for (int i = 0; i < 2; ++i) {
    int c = t * 2 + i;        // 0..255
    int row = c >> 2;         // 0..63
    int cb = (c & 3) * 16;    // byte offset within 64B row
    const char* g = (const char*)(X + (size_t)(row0 + row) * ld + k0) + cb;
    unsigned lo = (unsigned)(size_t)((const char*)sh + row * 64 + cb);
    __builtin_prefetch(g + 64, 0, 3);  // global_prefetch_b8, near scope
    async_ld_b128(lo, g);
  }
}

// ------------------------------------------------- CDNA5 Tensor Data Mover

// One wave DMAs a rows x cols (bf16) tile from global (row stride ld elems)
// into LDS via the TDM. D# built in SGPRs (groups 0..3, 2D tile, no padding,
// no multicast). Tracked by TENSORcnt.
DEV void tdm_load_tile(const u16* gsrc, unsigned ld, unsigned lds_byte,
                       unsigned rows, unsigned cols) {
  unsigned long long ga = (unsigned long long)(size_t)gsrc;
  const unsigned td0 = 1u << 20, td1 = 1u << 20;  // generous tensor dims (OOB off)
  v4u g0;
  g0[0] = 1u;                                   // count=1 valid descriptor
  g0[1] = lds_byte;                             // lds_addr
  g0[2] = (unsigned)ga;                         // global_addr[31:0]
  g0[3] = ((unsigned)(ga >> 32) & 0x1FFFFFFu)   // global_addr[56:32]
          | (2u << 30);                         // type = 2 ("image")
  v8u g1;
  g1[0] = 1u << 16;                             // data_size=1 (2B), mask=0
  g1[1] = (td0 & 0xFFFFu) << 16;                // tensor_dim0[15:0]
  g1[2] = (td0 >> 16) | ((td1 & 0xFFFFu) << 16);// tensor_dim0[31:16],dim1[15:0]
  g1[3] = (td1 >> 16) | (cols << 16);           // tensor_dim1[31:16],tile_dim0
  g1[4] = rows;                                 // tile_dim1 (tile_dim2=0: 2D)
  g1[5] = ld;                                   // tensor_dim0_stride[31:0]
  g1[6] = 0u;                                   // stride hi, dim1_stride lo
  g1[7] = 0u;
  v4u g2 = {0u, 0u, 0u, 0u};                    // dims 2/3 unused (tile_dim2=0)
  v4u g3 = {0u, 0u, 0u, 0u};
  asm volatile("tensor_load_to_lds %0, %1, %2, %3"
               :: "s"(g0), "s"(g1), "s"(g2), "s"(g3)
               : "memory");
}

// ---------------------------------------------------------------- utilities

__global__ void k_cvt_bf16(const float* __restrict__ in, u16* __restrict__ out,
                           long long n) {
  long long i = (long long)blockIdx.x * blockDim.x + threadIdx.x;
  long long stride = (long long)gridDim.x * blockDim.x;
  for (; i < n; i += stride) out[i] = f2bf(in[i]);
}

__global__ void k_embed(const int* __restrict__ x, const float* __restrict__ emb,
                        float* __restrict__ h) {
  int bl = blockIdx.x;
  int t = threadIdx.x;  // 256 threads, 4 floats each (D=1024)
  int tok = x[bl];
  *(float4*)(h + (size_t)bl * D_ + t * 4) =
      *(const float4*)(emb + (size_t)tok * D_ + t * 4);
}

__global__ void k_ln(const float* __restrict__ h, const float* __restrict__ w,
                     const float* __restrict__ bb, u16* __restrict__ xn) {
  int row = blockIdx.x;
  int t = threadIdx.x;
  float4 v = *(const float4*)(h + (size_t)row * D_ + t * 4);
  float s = v.x + v.y + v.z + v.w;
  float s2 = v.x * v.x + v.y * v.y + v.z * v.z + v.w * v.w;
  __shared__ float rs[256], rq[256];
  rs[t] = s; rq[t] = s2;
  __syncthreads();
  for (int k = 128; k > 0; k >>= 1) {
    if (t < k) { rs[t] += rs[t + k]; rq[t] += rq[t + k]; }
    __syncthreads();
  }
  float mu = rs[0] * (1.f / D_);
  float var = rq[0] * (1.f / D_) - mu * mu;
  float rinv = rsqrtf(var + 1e-5f);
  float4 wv = *(const float4*)(w + t * 4);
  float4 bv = *(const float4*)(bb + t * 4);
  u16 o0 = f2bf((v.x - mu) * rinv * wv.x + bv.x);
  u16 o1 = f2bf((v.y - mu) * rinv * wv.y + bv.y);
  u16 o2 = f2bf((v.z - mu) * rinv * wv.z + bv.z);
  u16 o3 = f2bf((v.w - mu) * rinv * wv.w + bv.w);
  uint2 pk;
  pk.x = (unsigned)o0 | ((unsigned)o1 << 16);
  pk.y = (unsigned)o2 | ((unsigned)o3 << 16);
  *(uint2*)(xn + (size_t)row * D_ + t * 4) = pk;
}

// ---------------------------------------------------------------- GEMM core

// Double-buffered 64x64 WG tile, 4 waves of 32x32.
// A tile staged by all waves via async global->LDS; B (weight) tile DMAed by
// wave 0 via the Tensor Data Mover. Both drained per-iteration, then barrier.
DEV void gemm_core(const u16* A, const u16* W, int m0wg, int n0wg, int lane,
                   int w, int t, u16* shA, u16* shB, v8f cc[4]) {
  bool tdmw = (t < 32);  // wave 0 owns the TDM
  issue_tile(A, D_, m0wg, 0, shA, t);
  if (tdmw)
    tdm_load_tile(W + (size_t)n0wg * D_, D_, (unsigned)(size_t)shB, 64, 32);
  wait_async0();
  if (tdmw) wait_tensor0();
  __syncthreads();
  int cur = 0;
  for (int k0 = 0; k0 < D_; k0 += 32) {
    int nxt = cur ^ 1;
    if (k0 + 32 < D_) {
      issue_tile(A, D_, m0wg, k0 + 32, shA + nxt * (64 * 32), t);
      if (tdmw)
        tdm_load_tile(W + (size_t)n0wg * D_ + (k0 + 32), D_,
                      (unsigned)(size_t)(shB + nxt * (64 * 32)), 64, 32);
    }
    const u16* sa = shA + cur * (64 * 32);
    const u16* sb = shB + cur * (64 * 32);
    v16bf a0 = load_frag_a(sa, 32, (w >> 1) * 32, 0, lane);
    v16bf a1 = load_frag_a(sa, 32, (w >> 1) * 32 + 16, 0, lane);
    v16bf b0 = load_frag_b(sb, 32, (w & 1) * 32, 0, lane);
    v16bf b1 = load_frag_b(sb, 32, (w & 1) * 32 + 16, 0, lane);
    cc[0] = wmma_bf16(a0, b0, cc[0]);
    cc[1] = wmma_bf16(a0, b1, cc[1]);
    cc[2] = wmma_bf16(a1, b0, cc[2]);
    cc[3] = wmma_bf16(a1, b1, cc[3]);
    wait_async0();            // next A tile resident
    if (tdmw) wait_tensor0(); // next B tile resident
    __syncthreads();          // all waves done reading `cur` before reuse
    cur = nxt;
  }
}

// C[M,3D] = xn[M,D] @ wx^T + bias, scattered into Q/K/V^T head layouts (bf16).
__global__ void __launch_bounds__(128) k_gemm_qkv(
    const u16* __restrict__ A, const u16* __restrict__ W,
    const float* __restrict__ bias, u16* __restrict__ Q, u16* __restrict__ Kb,
    u16* __restrict__ Vt) {
  int t = threadIdx.x, lane = t & 31, w = t >> 5;
  int m0wg = blockIdx.y * 64, n0wg = blockIdx.x * 64;
  __shared__ __align__(16) u16 shA[2 * 64 * 32];
  __shared__ __align__(16) u16 shB[2 * 64 * 32];
  v8f cc[4] = {v8zero(), v8zero(), v8zero(), v8zero()};
  gemm_core(A, W, m0wg, n0wg, lane, w, t, shA, shB, cc);

  int m0 = m0wg + (w >> 1) * 32, n0 = n0wg + (w & 1) * 32;
  int hf = lane >> 4, col = lane & 15;
  for (int ti = 0; ti < 2; ++ti)
    for (int tj = 0; tj < 2; ++tj) {
      v8f acc = cc[ti * 2 + tj];
      for (int r = 0; r < 8; ++r) {
        int m = m0 + ti * 16 + r + 8 * hf;
        int n = n0 + tj * 16 + col;
        u16 bvv = f2bf(acc[r] + bias[n]);
        int which = n >> 10, dcol = n & 1023;
        int head = dcol >> 6, c = dcol & 63;
        int b = m >> 11, l = m & 2047;  // L = 2048
        if (which == 0)
          Q[(((size_t)b * NH_ + head) * L_ + l) * DH_ + c] = bvv;
        else if (which == 1)
          Kb[(((size_t)b * NH_ + head) * L_ + l) * DH_ + c] = bvv;
        else
          Vt[(((size_t)b * NH_ + head) * DH_ + c) * L_ + l] = bvv;
      }
    }
}

// h = gelu(h + merged[M,D] @ wo^T + bias)  (exact gelu, fp32 residual in place)
__global__ void __launch_bounds__(128) k_gemm_out(
    const u16* __restrict__ A, const u16* __restrict__ W,
    const float* __restrict__ bias, float* __restrict__ h) {
  int t = threadIdx.x, lane = t & 31, w = t >> 5;
  int m0wg = blockIdx.y * 64, n0wg = blockIdx.x * 64;
  __shared__ __align__(16) u16 shA[2 * 64 * 32];
  __shared__ __align__(16) u16 shB[2 * 64 * 32];
  v8f cc[4] = {v8zero(), v8zero(), v8zero(), v8zero()};
  gemm_core(A, W, m0wg, n0wg, lane, w, t, shA, shB, cc);

  int m0 = m0wg + (w >> 1) * 32, n0 = n0wg + (w & 1) * 32;
  int hf = lane >> 4, col = lane & 15;
  for (int ti = 0; ti < 2; ++ti)
    for (int tj = 0; tj < 2; ++tj) {
      v8f acc = cc[ti * 2 + tj];
      for (int r = 0; r < 8; ++r) {
        int m = m0 + ti * 16 + r + 8 * hf;
        int n = n0 + tj * 16 + col;
        size_t idx = (size_t)m * D_ + n;
        float xx = h[idx] + acc[r] + bias[n];
        h[idx] = 0.5f * xx * (1.f + erff(xx * 0.70710678118654752f));
      }
    }
}

// ---------------------------------------------------------------- attention

// Flash attention. Grid (L/64, NH, B), 4 waves/WG; wave w owns 16 query rows.
__global__ void __launch_bounds__(128) k_attn(const u16* __restrict__ Q,
                                              const u16* __restrict__ Kb,
                                              const u16* __restrict__ Vt,
                                              u16* __restrict__ merged) {
  int lane = threadIdx.x & 31, w = threadIdx.x >> 5;
  int hh = blockIdx.y, b = blockIdx.z;
  int q0 = blockIdx.x * 64 + w * 16;
  const u16* Qh = Q + (((size_t)b * NH_ + hh) * L_) * DH_;
  const u16* Kh = Kb + (((size_t)b * NH_ + hh) * L_) * DH_;
  const u16* Vh = Vt + (((size_t)b * NH_ + hh) * DH_) * L_;
  __shared__ __align__(16) u16 psh[4][16 * 32];  // per-wave P staging (bf16)

  v16bf aq0 = load_frag_a(Qh, DH_, q0, 0, lane);
  v16bf aq1 = load_frag_a(Qh, DH_, q0, 32, lane);
  v8f o0 = v8zero(), o1 = v8zero(), o2 = v8zero(), o3 = v8zero();
  float mrun[8], srun[8];
  for (int r = 0; r < 8; ++r) { mrun[r] = NEG_INF; srun[r] = 0.f; }
  int hf = lane >> 4, col = lane & 15;
  const float scale = 0.03125f;  // 1/sqrt(D) — full model dim, as in reference

  for (int j0 = 0; j0 <= q0 + 15; j0 += 32) {  // wave-uniform causal bound
    v8f s0 = v8zero(), s1 = v8zero();
    s0 = wmma_bf16(aq0, load_frag_b(Kh, DH_, j0, 0, lane), s0);
    s0 = wmma_bf16(aq1, load_frag_b(Kh, DH_, j0, 32, lane), s0);
    s1 = wmma_bf16(aq0, load_frag_b(Kh, DH_, j0 + 16, 0, lane), s1);
    s1 = wmma_bf16(aq1, load_frag_b(Kh, DH_, j0 + 16, 32, lane), s1);

    for (int r = 0; r < 8; ++r) {
      int qrow = q0 + r + 8 * hf;
      float e0 = (j0 + col <= qrow) ? s0[r] * scale : NEG_INF;
      float e1 = (j0 + 16 + col <= qrow) ? s1[r] * scale : NEG_INF;
      float rm = fmaxf(e0, e1);  // row max across the 16-lane half
      rm = fmaxf(rm, __shfl_xor(rm, 1));
      rm = fmaxf(rm, __shfl_xor(rm, 2));
      rm = fmaxf(rm, __shfl_xor(rm, 4));
      rm = fmaxf(rm, __shfl_xor(rm, 8));
      float mnew = fmaxf(mrun[r], rm);
      float alpha = expf(mrun[r] - mnew);
      mrun[r] = mnew;
      float p0 = expf(e0 - mnew), p1 = expf(e1 - mnew);
      float ps = p0 + p1;
      ps += __shfl_xor(ps, 1);
      ps += __shfl_xor(ps, 2);
      ps += __shfl_xor(ps, 4);
      ps += __shfl_xor(ps, 8);
      srun[r] = srun[r] * alpha + ps;
      o0[r] *= alpha; o1[r] *= alpha; o2[r] *= alpha; o3[r] *= alpha;
      psh[w][(r + 8 * hf) * 32 + col] = f2bf(p0);
      psh[w][(r + 8 * hf) * 32 + col + 16] = f2bf(p1);
    }
    v16bf pa = load_frag_a(&psh[w][0], 32, 0, 0, lane);  // P from LDS (ds_load)
    o0 = wmma_bf16(pa, load_frag_b(Vh, L_, 0, j0, lane), o0);
    o1 = wmma_bf16(pa, load_frag_b(Vh, L_, 16, j0, lane), o1);
    o2 = wmma_bf16(pa, load_frag_b(Vh, L_, 32, j0, lane), o2);
    o3 = wmma_bf16(pa, load_frag_b(Vh, L_, 48, j0, lane), o3);
  }

  for (int r = 0; r < 8; ++r) {
    float inv = 1.f / srun[r];
    int qrow = q0 + r + 8 * hf;
    size_t base = ((size_t)b * L_ + qrow) * D_ + (size_t)hh * DH_;
    merged[base + col] = f2bf(o0[r] * inv);
    merged[base + 16 + col] = f2bf(o1[r] * inv);
    merged[base + 32 + col] = f2bf(o2[r] * inv);
    merged[base + 48 + col] = f2bf(o3[r] * inv);
  }
}

// ---------------------------------------------------------------- head

__global__ void k_hlast(const float* __restrict__ h, u16* __restrict__ hl) {
  int i = blockIdx.x * 256 + threadIdx.x;  // 16 rows (4 valid) x 1024
  int row = i >> 10, c = i & 1023;
  float v = (row < B_) ? h[((size_t)(row * L_ + (L_ - 1))) * D_ + c] : 0.f;
  hl[i] = f2bf(v);
}

__global__ void __launch_bounds__(32) k_logits(const u16* __restrict__ hlast,
                                               const u16* __restrict__ actor,
                                               const float* __restrict__ ab,
                                               float* __restrict__ logits) {
  int lane = threadIdx.x & 31;
  int n0 = blockIdx.x * 16;
  v8f c = v8zero();
  for (int k0 = 0; k0 < D_; k0 += 32) {
    v16bf a = load_frag_a(hlast, D_, 0, k0, lane);
    v16bf bb = load_frag_b(actor, D_, n0, k0, lane);
    c = wmma_bf16(a, bb, c);
  }
  int hf = lane >> 4, col = lane & 15;
  for (int r = 0; r < 8; ++r) {
    int m = r + 8 * hf;
    if (m < B_) logits[(size_t)m * V_ + n0 + col] = c[r] + ab[n0 + col];
  }
}

__global__ void k_softmax(const float* __restrict__ logits,
                          float* __restrict__ out) {
  int b = blockIdx.x, t = threadIdx.x;
  const float* row = logits + (size_t)b * V_;
  __shared__ float red[256];
  float mx = NEG_INF;
  for (int i = t; i < V_; i += 256) mx = fmaxf(mx, row[i]);
  red[t] = mx;
  __syncthreads();
  for (int s = 128; s > 0; s >>= 1) {
    if (t < s) red[t] = fmaxf(red[t], red[t + s]);
    __syncthreads();
  }
  mx = red[0];
  __syncthreads();
  float sum = 0.f;
  for (int i = t; i < V_; i += 256) sum += expf(row[i] - mx);
  red[t] = sum;
  __syncthreads();
  for (int s = 128; s > 0; s >>= 1) {
    if (t < s) red[t] += red[t + s];
    __syncthreads();
  }
  float inv = 1.f / red[0];
  for (int i = t; i < V_; i += 256) out[(size_t)b * V_ + i] = expf(row[i] - mx) * inv;
}

// ---------------------------------------------------------------- host

extern "C" void kernel_launch(void* const* d_in, const int* in_sizes, int n_in,
                              void* d_out, int out_size, void* d_ws,
                              size_t ws_size, hipStream_t stream) {
  (void)in_sizes; (void)n_in; (void)out_size; (void)ws_size;
  const int*   x       = (const int*)d_in[0];
  /* d_in[1] = nh scalar, constant NH_ */
  const float* emb     = (const float*)d_in[2];
  const float* wx_w    = (const float*)d_in[3];
  const float* wx_b    = (const float*)d_in[4];
  const float* wo_w    = (const float*)d_in[5];
  const float* wo_b    = (const float*)d_in[6];
  const float* ln_w    = (const float*)d_in[7];
  const float* ln_b    = (const float*)d_in[8];
  const float* actor_w = (const float*)d_in[9];
  const float* actor_b = (const float*)d_in[10];

  char* ws = (char*)d_ws;
  constexpr size_t SZ_H   = (size_t)B_ * L_ * D_ * 4;
  constexpr size_t SZ_BF  = (size_t)B_ * L_ * D_ * 2;
  constexpr size_t OFF_H      = 0;
  constexpr size_t OFF_XN     = OFF_H + SZ_H;
  constexpr size_t OFF_Q      = OFF_XN + SZ_BF;
  constexpr size_t OFF_K      = OFF_Q + SZ_BF;
  constexpr size_t OFF_VT     = OFF_K + SZ_BF;
  constexpr size_t OFF_MERGED = OFF_VT + SZ_BF;
  constexpr size_t OFF_WX     = OFF_MERGED + SZ_BF;
  constexpr size_t OFF_WO     = OFF_WX + (size_t)NL_ * D3_ * D_ * 2;
  constexpr size_t OFF_ACT    = OFF_WO + (size_t)NL_ * D_ * D_ * 2;
  constexpr size_t OFF_HL     = OFF_ACT + (size_t)V_ * D_ * 2;
  constexpr size_t OFF_LOG    = OFF_HL + (size_t)16 * D_ * 2;

  float* hbuf   = (float*)(ws + OFF_H);
  u16*   xn     = (u16*)(ws + OFF_XN);
  u16*   Qb     = (u16*)(ws + OFF_Q);
  u16*   Kbuf   = (u16*)(ws + OFF_K);
  u16*   Vtb    = (u16*)(ws + OFF_VT);
  u16*   merged = (u16*)(ws + OFF_MERGED);
  u16*   wx16   = (u16*)(ws + OFF_WX);
  u16*   wo16   = (u16*)(ws + OFF_WO);
  u16*   act16  = (u16*)(ws + OFF_ACT);
  u16*   hl16   = (u16*)(ws + OFF_HL);
  float* logits = (float*)(ws + OFF_LOG);

  k_cvt_bf16<<<4096, 256, 0, stream>>>(wx_w, wx16, (long long)NL_ * D3_ * D_);
  k_cvt_bf16<<<4096, 256, 0, stream>>>(wo_w, wo16, (long long)NL_ * D_ * D_);
  k_cvt_bf16<<<8192, 256, 0, stream>>>(actor_w, act16, (long long)V_ * D_);
  k_embed<<<B_ * L_, 256, 0, stream>>>(x, emb, hbuf);

  for (int layer = 0; layer < NL_; ++layer) {
    k_ln<<<B_ * L_, 256, 0, stream>>>(hbuf, ln_w + (size_t)layer * D_,
                                      ln_b + (size_t)layer * D_, xn);
    k_gemm_qkv<<<dim3(D3_ / 64, (B_ * L_) / 64), 128, 0, stream>>>(
        xn, wx16 + (size_t)layer * D3_ * D_, wx_b + (size_t)layer * D3_, Qb,
        Kbuf, Vtb);
    k_attn<<<dim3(L_ / 64, NH_, B_), 128, 0, stream>>>(Qb, Kbuf, Vtb, merged);
    k_gemm_out<<<dim3(D_ / 64, (B_ * L_) / 64), 128, 0, stream>>>(
        merged, wo16 + (size_t)layer * D_ * D_, wo_b + (size_t)layer * D_,
        hbuf);
  }

  k_hlast<<<64, 256, 0, stream>>>(hbuf, hl16);
  k_logits<<<V_ / 16, 32, 0, stream>>>(hl16, act16, actor_b, logits);
  k_softmax<<<B_, 256, 0, stream>>>(logits, (float*)d_out);
}
